// Attention_52553219834371
// MI455X (gfx1250) — matrix operand order repaired and checked by reading, not compiled, not run on hardware
//
#include <hip/hip_runtime.h>

// ---------------- problem constants ----------------
#define B_   16
#define N_   4096
#define C_   1024
#define H_   16
#define HD_  64
#define QL_  64
#define SCALE_ 0.125f   // 64^-0.5

// ---------------- bf16 WMMA plumbing ----------------
typedef __bf16 bf16_t;
typedef __attribute__((ext_vector_type(16))) __bf16 v16bf;
typedef __attribute__((ext_vector_type(8)))  __bf16 v8bf;
typedef __attribute__((ext_vector_type(8)))  float  v8f;
typedef __attribute__((ext_vector_type(4)))  int    v4i;

__device__ __forceinline__ v8f v8f_zero() {
  v8f z = {0.f,0.f,0.f,0.f,0.f,0.f,0.f,0.f};
  return z;
}

__device__ __forceinline__ v8f wmma_bf16(v16bf a, v16bf b, v8f c) {
  // D = A(16x32 bf16) x B(32x16 bf16) + C(16x16 f32)
  return __builtin_amdgcn_wmma_f32_16x16x32_bf16(
      /*neg_a=*/false, a, /*neg_b=*/false, b,
      /*c_mod=*/(short)0, c, /*reuse_a=*/false, /*reuse_b=*/false);
}

// fp32 -> bf16 (round to nearest even), returned as raw bits
__device__ __forceinline__ unsigned short f2bf(float f) {
  unsigned int u = __float_as_uint(f);
  u += 0x7FFFu + ((u >> 16) & 1u);
  return (unsigned short)(u >> 16);
}

// A-matrix (16x32, bf16) fragment load for one lane.
// k order used here (same as ldb32x16): lane row = lane%16; half=lane/16; elems 0..15 -> K=16*half+e.
// `p` points at row start (K-contiguous).
__device__ __forceinline__ v16bf lda16x32(const unsigned short* p, int half) {
  v8bf lo = *(const v8bf*)(p + half * 16);
  v8bf hi = *(const v8bf*)(p + half * 16 + 8);
  v16bf r;
#pragma unroll
  for (int i = 0; i < 8; ++i) { r[i] = lo[i]; r[8 + i] = hi[i]; }
  return r;
}

// B-matrix (32x16, bf16) fragment: lane col = lane%16, elems K = 16*half + e.
// `rowp` points at element (K=0) of this lane's column stream (K-contiguous).
__device__ __forceinline__ v16bf ldb32x16(const unsigned short* rowp, int half) {
  return *(const v16bf*)(rowp + half * 16);
}

// ---------------- CDNA5 async copy: global -> LDS (16 B per lane) ----------------
// Tracked by ASYNCcnt; no VGPR data round-trip. The builtin takes int4* operands:
// (global src, lds dst, imm offset, cpol).
__device__ __forceinline__ void async_copy16(const void* gsrc, void* lds_dst) {
#if __has_builtin(__builtin_amdgcn_global_load_async_to_lds_b128)
  __builtin_amdgcn_global_load_async_to_lds_b128(
      (__attribute__((address_space(1))) v4i*)gsrc,
      (__attribute__((address_space(3))) v4i*)lds_dst, 0, 0);
#else
  asm volatile("global_load_async_to_lds_b128 %0, %1, off"
               :: "v"((unsigned int)(unsigned long long)lds_dst),
                  "v"(gsrc)
               : "memory");
#endif
}

__device__ __forceinline__ void wait_async0() {
#if __has_builtin(__builtin_amdgcn_s_wait_asynccnt)
  __builtin_amdgcn_s_wait_asynccnt(0);
#else
  asm volatile("s_wait_asynccnt 0x0" ::: "memory");
#endif
}

// ---------------- kernel 1: fp32 -> bf16 convert ----------------
__global__ void cvt_f32_bf16(const float* __restrict__ in,
                             unsigned short* __restrict__ out,
                             long long n4) {  // number of float4 groups
  long long i = (long long)blockIdx.x * blockDim.x + threadIdx.x;
  long long stride = (long long)gridDim.x * blockDim.x;
  for (; i < n4; i += stride) {
    float4 v = ((const float4*)in)[i];
    unsigned long long p =
        (unsigned long long)f2bf(v.x) |
        ((unsigned long long)f2bf(v.y) << 16) |
        ((unsigned long long)f2bf(v.z) << 32) |
        ((unsigned long long)f2bf(v.w) << 48);
    ((unsigned long long*)out)[i] = p;
  }
}

// ---------------- kernel 2: generic bf16 WMMA GEMM ----------------
// D[m,n] = scale * sum_k A[m,k] * Bt[n,k]  (+ bias[n])
// qmode==1: logical row m maps to x-row (m/64)*4096 + (m%64) (first-64-token Q proj)
__global__ void gemm_bf16_bt(const unsigned short* __restrict__ A,
                             const unsigned short* __restrict__ Bt,
                             int M, int N, int K, int qmode, float scale,
                             const float* __restrict__ bias,
                             float* __restrict__ outF,
                             unsigned short* __restrict__ outB) {
  const int lane = threadIdx.x & 31;
  const int wave = threadIdx.x >> 5;
  const int wpb  = blockDim.x >> 5;
  const int col  = lane & 15;
  const int half = lane >> 4;
  const int tilesM = M >> 4, tilesN = N >> 4;
  const int nTiles = tilesM * tilesN;

  for (int t = blockIdx.x * wpb + wave; t < nTiles; t += gridDim.x * wpb) {
    const int tm = t / tilesN, tn = t % tilesN;
    const int m0 = tm << 4, n0 = tn << 4;

    const int am = m0 + col;  // this lane's A row
    const long long arow = qmode ? ((long long)(am >> 6) * N_ + (am & 63))
                                 : (long long)am;
    const unsigned short* ap = A + arow * K;
    const unsigned short* bp = Bt + (long long)(n0 + col) * K;

    v8f acc = v8f_zero();
    for (int k0 = 0; k0 < K; k0 += 32) {
      v16bf a = lda16x32(ap + k0, half);
      v16bf b = ldb32x16(bp + k0, half);
      acc = wmma_bf16(a, b, acc);
    }

    const float bv = bias ? bias[n0 + col] : 0.f;
#pragma unroll
    for (int v = 0; v < 8; ++v) {
      const int m = m0 + v + 8 * half;
      const float val = acc[v] * scale + bv;
      const long long idx = (long long)m * N + n0 + col;
      if (outF) outF[idx] = val;
      if (outB) outB[idx] = f2bf(val);
    }
  }
}

// ---------------- kernel 3: fused K/V-projection + flash attention ----------------
// One block per (b,h). Per 128-key chunk: all 8 waves project K/V via WMMA with
// Wk/Wv k-tiles async-staged in LDS (shared by the whole block); waves 0..3 then
// run S = Q K^T, online softmax, O += P V — all on WMMA.
#define CHUNK 128
#define KT    64    // k-tile staged per async round
#define LDW   72    // staged weight row stride (padded)
#define LDK   72    // Ks row stride (keys x d), padded
#define LDVT  136   // Vts row stride (d x keys), padded
#define LDP   136   // per-wave P strip stride
#define LDQ   72    // Qs row stride

__global__ void __launch_bounds__(256)
fused_flash_attn(const unsigned short* __restrict__ xb,   // [B,N,C] bf16
                 const unsigned short* __restrict__ qb,   // [B,64,C] bf16, pre-scaled
                 const unsigned short* __restrict__ wkb,  // [C,C] bf16 (row = out-ch)
                 const unsigned short* __restrict__ wvb,  // [C,C] bf16
                 unsigned short* __restrict__ ab) {       // [B,64,C] bf16 out
  __shared__ unsigned short Qs[QL_ * LDQ];        //  9 KB
  __shared__ unsigned short Ks[CHUNK * LDK];      // 18 KB  (key-major, d-contig)
  __shared__ unsigned short Vts[HD_ * LDVT];      // 17 KB  (d-major, key-contig)
  // union: phase 1 = Wk/Wv k-tile staging (2 * 64 * 72), phase 2 = P strips
  __shared__ unsigned short Scratch[2 * HD_ * LDW];  // 18 KB

  unsigned short* Wk_s = &Scratch[0];
  unsigned short* Wv_s = &Scratch[HD_ * LDW];

  const int bh = blockIdx.x;
  const int b = bh >> 4, h = bh & 15;
  const int tid = threadIdx.x;
  const int lane = tid & 31, wave = tid >> 5;
  const int col = lane & 15, half = lane >> 4;

  // stage Q strip (64 x 64 bf16) into LDS
  {
    const int r = tid >> 2;            // 0..63
    const int c0 = (tid & 3) * 16;     // 0,16,32,48
    const unsigned short* src = qb + ((long long)b * QL_ + r) * C_ + h * HD_ + c0;
    *(v8bf*)(&Qs[r * LDQ + c0])     = *(const v8bf*)(src);
    *(v8bf*)(&Qs[r * LDQ + c0 + 8]) = *(const v8bf*)(src + 8);
  }

  // flash state for waves 0..3 (row = v + 8*half within this wave's 16-q strip)
  v8f o0 = v8f_zero(), o1 = v8f_zero(), o2 = v8f_zero(), o3 = v8f_zero();
  float mrow[8], lrow[8];
#pragma unroll
  for (int v = 0; v < 8; ++v) { mrow[v] = -3.0e38f; lrow[v] = 0.f; }

  const unsigned short* wkh = wkb + (long long)h * HD_ * C_;
  const unsigned short* wvh = wvb + (long long)h * HD_ * C_;

  // cooperative async-staging coordinates: 512 16B units per weight tile,
  // 2 units per thread; unit u -> row u/8, ushort col (u%8)*8
  const int u0 = tid * 2;
  const int sr0 = u0 >> 3,       sc0 = (u0 & 7) * 8;
  const int sr1 = (u0 + 1) >> 3, sc1 = ((u0 + 1) & 7) * 8;

  for (int n0 = 0; n0 < N_; n0 += CHUNK) {
    __syncthreads();  // LDS safe to (re)write (Scratch was P in prior phase 2)

    const unsigned short* xrow =
        xb + ((long long)b * N_ + n0 + wave * 16 + col) * C_;
    if (n0 + CHUNK < N_)  // hint next chunk's x rows toward the caches
      __builtin_prefetch(xrow + (long long)CHUNK * C_, 0, 1);

    // ---- phase 1: all 8 waves project this chunk's K and V (M=16 keys/wave),
    //      Wk/Wv staged per k-tile via async global->LDS copies ----
    v8f kacc[4], vacc[4];
#pragma unroll
    for (int j = 0; j < 4; ++j) { kacc[j] = v8f_zero(); vacc[j] = v8f_zero(); }

    for (int kt = 0; kt < C_ / KT; ++kt) {
      // async-stage Wk/Wv [64 out-ch x 64 k] tiles (16 B per copy, ASYNCcnt-tracked)
      const long long g0 = (long long)sr0 * C_ + kt * KT + sc0;
      const long long g1 = (long long)sr1 * C_ + kt * KT + sc1;
      async_copy16(wkh + g0, &Wk_s[sr0 * LDW + sc0]);
      async_copy16(wkh + g1, &Wk_s[sr1 * LDW + sc1]);
      async_copy16(wvh + g0, &Wv_s[sr0 * LDW + sc0]);
      async_copy16(wvh + g1, &Wv_s[sr1 * LDW + sc1]);
      wait_async0();
      __syncthreads();

#pragma unroll
      for (int kk = 0; kk < 2; ++kk) {  // two 32-wide K steps per tile
        v16bf a = lda16x32(xrow + kt * KT + kk * 32, half);
#pragma unroll
        for (int j = 0; j < 4; ++j) {
          v16bf bk = ldb32x16(&Wk_s[(j * 16 + col) * LDW + kk * 32], half);
          kacc[j] = wmma_bf16(a, bk, kacc[j]);
          v16bf bv = ldb32x16(&Wv_s[(j * 16 + col) * LDW + kk * 32], half);
          vacc[j] = wmma_bf16(a, bv, vacc[j]);
        }
      }
      __syncthreads();  // staging tile consumed; safe to overwrite next kt
    }

    // spill to LDS: Ks[key][d] (S-phase B operand), Vts[d][key] (O-phase B operand)
#pragma unroll
    for (int j = 0; j < 4; ++j) {
#pragma unroll
      for (int v = 0; v < 8; ++v) {
        const int keyl = wave * 16 + v + 8 * half;
        const int d = j * 16 + col;
        Ks[keyl * LDK + d]   = f2bf(kacc[j][v]);
        Vts[d * LDVT + keyl] = f2bf(vacc[j][v]);
      }
    }
    __syncthreads();

    // ---- phase 2: waves 0..3 — attention over this chunk ----
    if (wave < 4) {
      // S strip: 16 queries (this wave) x 128 keys = 8 D-tiles
      v8f s[8];
#pragma unroll
      for (int j = 0; j < 8; ++j) s[j] = v8f_zero();

      const unsigned short* qrow = &Qs[(wave * 16 + col) * LDQ];
#pragma unroll
      for (int kk = 0; kk < 2; ++kk) {  // head-dim in chunks of 32
        v16bf a = lda16x32(qrow + kk * 32, half);
#pragma unroll
        for (int j = 0; j < 8; ++j) {
          v16bf bkt = ldb32x16(&Ks[(j * 16 + col) * LDK + kk * 32], half);
          s[j] = wmma_bf16(a, bkt, s[j]);
        }
      }

      // online softmax; row (v+8*half) is spread over 16 lanes of one half
      float cm[8];
#pragma unroll
      for (int v = 0; v < 8; ++v) {
        float x = s[0][v];
#pragma unroll
        for (int j = 1; j < 8; ++j) x = fmaxf(x, s[j][v]);
#pragma unroll
        for (int m = 1; m < 16; m <<= 1) x = fmaxf(x, __shfl_xor(x, m, 32));
        cm[v] = x;
      }
      float rs[8];
#pragma unroll
      for (int v = 0; v < 8; ++v) {
        const float mnew = fmaxf(mrow[v], cm[v]);
        const float corr = __expf(mrow[v] - mnew);
        mrow[v] = mnew;
        lrow[v] *= corr;
        o0[v] *= corr; o1[v] *= corr; o2[v] *= corr; o3[v] *= corr;
        float sum = 0.f;
#pragma unroll
        for (int j = 0; j < 8; ++j) {
          const float p = __expf(s[j][v] - mnew);
          s[j][v] = p;
          sum += p;
        }
        rs[v] = sum;
      }
#pragma unroll
      for (int v = 0; v < 8; ++v) {
        float x = rs[v];
#pragma unroll
        for (int m = 1; m < 16; m <<= 1) x += __shfl_xor(x, m, 32);
        lrow[v] += x;
      }

      // transpose P (D-layout) -> A-layout via this wave's private LDS strip
      unsigned short* Pw = &Scratch[wave * 16 * LDP];
#pragma unroll
      for (int j = 0; j < 8; ++j)
#pragma unroll
        for (int v = 0; v < 8; ++v)
          Pw[(v + 8 * half) * LDP + j * 16 + col] = f2bf(s[j][v]);

      // O += P @ V   (A = P[16 x 128], B = Vts columns, key-contiguous)
      const unsigned short* prow = &Pw[col * LDP];
#pragma unroll
      for (int kk = 0; kk < 4; ++kk) {  // keys in chunks of 32
        v16bf a = lda16x32(prow + kk * 32, half);
        o0 = wmma_bf16(a, ldb32x16(&Vts[( 0 + col) * LDVT + kk * 32], half), o0);
        o1 = wmma_bf16(a, ldb32x16(&Vts[(16 + col) * LDVT + kk * 32], half), o1);
        o2 = wmma_bf16(a, ldb32x16(&Vts[(32 + col) * LDVT + kk * 32], half), o2);
        o3 = wmma_bf16(a, ldb32x16(&Vts[(48 + col) * LDVT + kk * 32], half), o3);
      }
    }
  }

  // epilogue: normalize and write attention output as bf16 [B,64,C]
  if (wave < 4) {
#pragma unroll
    for (int v = 0; v < 8; ++v) {
      const float inv = 1.0f / lrow[v];
      const int q = wave * 16 + v + 8 * half;
      const long long base = ((long long)b * QL_ + q) * C_ + h * HD_;
      ab[base +  0 + col] = f2bf(o0[v] * inv);
      ab[base + 16 + col] = f2bf(o1[v] * inv);
      ab[base + 32 + col] = f2bf(o2[v] * inv);
      ab[base + 48 + col] = f2bf(o3[v] * inv);
    }
  }
}

// ---------------- host-side launch ----------------
extern "C" void kernel_launch(void* const* d_in, const int* in_sizes, int n_in,
                              void* d_out, int out_size, void* d_ws, size_t ws_size,
                              hipStream_t stream) {
  (void)in_sizes; (void)n_in; (void)out_size; (void)ws_size;
  const float* x  = (const float*)d_in[0];
  const float* Wq = (const float*)d_in[1];
  const float* Wk = (const float*)d_in[2];
  const float* Wv = (const float*)d_in[3];
  const float* Wp = (const float*)d_in[4];
  const float* bp = (const float*)d_in[5];
  float* out = (float*)d_out;

  // workspace layout (all bf16 as raw ushort)
  char* ws = (char*)d_ws;
  const size_t SZ_X = (size_t)B_ * N_ * C_ * 2;   // 128 MB
  const size_t SZ_W = (size_t)C_ * C_ * 2;        //   2 MB
  const size_t SZ_Q = (size_t)B_ * QL_ * C_ * 2;  //   2 MB
  unsigned short* xb  = (unsigned short*)(ws);
  unsigned short* wqb = (unsigned short*)(ws + SZ_X);
  unsigned short* wkb = (unsigned short*)(ws + SZ_X + SZ_W);
  unsigned short* wvb = (unsigned short*)(ws + SZ_X + 2 * SZ_W);
  unsigned short* wpb = (unsigned short*)(ws + SZ_X + 3 * SZ_W);
  unsigned short* qb  = (unsigned short*)(ws + SZ_X + 4 * SZ_W);
  unsigned short* ab  = (unsigned short*)(ws + SZ_X + 4 * SZ_W + SZ_Q);

  // 1) fp32 -> bf16 conversions (x pass makes later x reads L2-resident: 128MB < 192MB L2)
  cvt_f32_bf16<<<4096, 256, 0, stream>>>(x,  xb,  (long long)B_ * N_ * C_ / 4);
  cvt_f32_bf16<<<512,  256, 0, stream>>>(Wq, wqb, (long long)C_ * C_ / 4);
  cvt_f32_bf16<<<512,  256, 0, stream>>>(Wk, wkb, (long long)C_ * C_ / 4);
  cvt_f32_bf16<<<512,  256, 0, stream>>>(Wv, wvb, (long long)C_ * C_ / 4);
  cvt_f32_bf16<<<512,  256, 0, stream>>>(Wp, wpb, (long long)C_ * C_ / 4);

  // 2) Q projection (first 64 tokens/batch), softmax scale folded into Q
  gemm_bf16_bt<<<512, 256, 0, stream>>>(xb, wqb, B_ * QL_, C_, C_,
                                        /*qmode=*/1, SCALE_, nullptr,
                                        nullptr, qb);

  // 3) fused K/V projection + flash attention, one block per (b,h)
  fused_flash_attn<<<B_ * H_, 256, 0, stream>>>(xb, qb, wkb, wvb, ab);

  // 4) output projection + bias -> fp32 d_out
  gemm_bf16_bt<<<512, 256, 0, stream>>>(ab, wpb, B_ * QL_, C_, C_,
                                        /*qmode=*/0, 1.0f, bp,
                                        out, nullptr);
}
